// LSTM_59485297049830
// MI455X (gfx1250) — matrix operand order repaired and checked by reading, not compiled, not run on hardware
//
#include <hip/hip_runtime.h>

typedef __attribute__((ext_vector_type(16))) _Float16 v16h;
typedef __attribute__((ext_vector_type(8)))  float    v8f;
typedef __attribute__((ext_vector_type(4)))  float    v4f;

#define T_STEPS  262144
#define I_DIM    7
#define H_DIM    100
#define G_DIM    400     // 4*H
#define KP       128     // K padded to 4 chunks of 32
#define NTILES   25      // 400 / 16
#define NTHREADS 256
#define NWAVES   8
#define SLOTS    4       // tiles per wave (wave w handles tiles w, w+8, w+16, w+24)

__global__ __launch_bounds__(NTHREADS, 1)
void lstm_persistent_wmma(const float* __restrict__ x,
                          const float* __restrict__ W_ih,
                          const float* __restrict__ W_hh,
                          const float* __restrict__ b_ih,
                          const float* __restrict__ b_hh,
                          const float* __restrict__ W_lin,
                          const float* __restrict__ b_lin,
                          float* __restrict__ out)
{
    __shared__ __align__(32) _Float16 h16[KP];   // h in f16 (B operand), [100..127] = 0
    __shared__ float xs[I_DIM];                  // current x_t
    __shared__ __align__(16) float xg[G_DIM];    // input-side gate preactivation + biases
    __shared__ __align__(16) float garr[G_DIM];  // W_hh @ h (WMMA result)
    __shared__ float h32[H_DIM];                 // h in f32 (for final linear)

    const int tid  = threadIdx.x;
    const int lane = tid & 31;
    const int wave = tid >> 5;

    // ---- per-thread x-side weights: rows j0 = tid, j1 = tid + 256 ----
    float wih0[I_DIM], wih1[I_DIM];
    float bsum0 = 0.f, bsum1 = 0.f;
    const int j0 = tid, j1 = tid + NTHREADS;
    {
        #pragma unroll
        for (int i = 0; i < I_DIM; ++i) wih0[i] = W_ih[j0 * I_DIM + i];
        bsum0 = b_ih[j0] + b_hh[j0];
        if (j1 < G_DIM) {
            #pragma unroll
            for (int i = 0; i < I_DIM; ++i) wih1[i] = W_ih[j1 * I_DIM + i];
            bsum1 = b_ih[j1] + b_hh[j1];
        }
    }

    // ---- preload W_hh into WMMA A fragments (f16), resident for whole run ----
    // 16-bit A 16x32 layout: lane L row M=L%16; element e;
    // K = (L>=16 ? 16 : 0) + e  (contiguous, matches B fill)
    v16h afrag[SLOTS][4];
    const int m  = lane & 15;
    const int hi = (lane >> 4) & 1;
    #pragma unroll
    for (int s = 0; s < SLOTS; ++s) {
        const int tile = wave + NWAVES * s;
        #pragma unroll
        for (int kc = 0; kc < 4; ++kc) {
            v16h av;
            #pragma unroll
            for (int e = 0; e < 16; ++e) {
                const int K = (hi ? 16 : 0) + e;
                const int col = kc * 32 + K;
                float w = 0.f;
                if (tile < NTILES && col < H_DIM)
                    w = W_hh[(tile * 16 + m) * H_DIM + col];
                av[e] = (_Float16)w;
            }
            afrag[s][kc] = av;
        }
    }

    // ---- init state ----
    if (tid < KP)    h16[tid] = (_Float16)0.f;
    if (tid < I_DIM) xs[tid]  = x[tid];
    float c = 0.f;
    __syncthreads();

    const bool iolane = (lane & 15) == 0;   // lanes 0 and 16 carry column N=0
    const int  hb     = hi << 4;            // +16 halves for the upper K half

    for (int t = 0; t < T_STEPS; ++t) {
        // --- input-side preactivation (registers -> LDS), independent of WMMA phase ---
        {
            float s0 = bsum0;
            #pragma unroll
            for (int i = 0; i < I_DIM; ++i) s0 += xs[i] * wih0[i];
            xg[j0] = s0;
            if (j1 < G_DIM) {
                float s1 = bsum1;
                #pragma unroll
                for (int i = 0; i < I_DIM; ++i) s1 += xs[i] * wih1[i];
                xg[j1] = s1;
            }
        }

        // --- recurrent matvec garr = W_hh @ h via WMMA ---
        // B 32x16 layout: lane = column N; only N=0 used -> lanes 0/16 hold h.
        v16h b0 = {}, b1 = {}, b2 = {}, b3 = {};
        if (iolane) {
            b0 = *(const v16h*)(&h16[hb]);
            b1 = *(const v16h*)(&h16[32 + hb]);
            b2 = *(const v16h*)(&h16[64 + hb]);
            b3 = *(const v16h*)(&h16[96 + hb]);
        }
        #pragma unroll
        for (int s = 0; s < SLOTS; ++s) {
            const int tile = wave + NWAVES * s;   // wave-uniform
            if (tile >= NTILES) break;
            v8f acc = {};
            acc = __builtin_amdgcn_wmma_f32_16x16x32_f16(false, afrag[s][0], false, b0, (short)0, acc, false, false);
            acc = __builtin_amdgcn_wmma_f32_16x16x32_f16(false, afrag[s][1], false, b1, (short)0, acc, false, false);
            acc = __builtin_amdgcn_wmma_f32_16x16x32_f16(false, afrag[s][2], false, b2, (short)0, acc, false, false);
            acc = __builtin_amdgcn_wmma_f32_16x16x32_f16(false, afrag[s][3], false, b3, (short)0, acc, false, false);
            // D column 0: lane 0 holds M=0..7 (acc[0..7]), lane 16 holds M=8..15.
            // One predicate, address selected by hi -> two ds_store_b128 per tile.
            if (iolane) {
                v4f lo4 = { acc[0], acc[1], acc[2], acc[3] };
                v4f hi4 = { acc[4], acc[5], acc[6], acc[7] };
                v4f* p = (v4f*)(&garr[tile * 16 + (hi << 3)]);
                p[0] = lo4;
                p[1] = hi4;
            }
        }
        __syncthreads();

        // --- gate activations + state update (threads 0..99) ---
        if (tid < H_DIM) {
            const float gi = xg[tid]             + garr[tid];
            const float gf = xg[H_DIM + tid]     + garr[H_DIM + tid];
            const float gc = xg[2 * H_DIM + tid] + garr[2 * H_DIM + tid];
            const float go = xg[3 * H_DIM + tid] + garr[3 * H_DIM + tid];
            const float ig = 1.f / (1.f + __expf(-gi));
            const float fg = 1.f / (1.f + __expf(-gf));
            const float cg = tanhf(gc);
            const float og = 1.f / (1.f + __expf(-go));
            c = fg * c + ig * cg;
            const float h = og * tanhf(c);
            h32[tid] = h;
            h16[tid] = (_Float16)h;
        }
        // prefetch next x_t (28 bytes; input stays hot in L2)
        if (tid < I_DIM && (t + 1) < T_STEPS) xs[tid] = x[(t + 1) * I_DIM + tid];
        __syncthreads();
    }

    // ---- final linear: pred = relu(h_T) @ W_lin^T + b_lin (O = 1) ----
    if (tid == 0) {
        float acc = b_lin[0];
        for (int j = 0; j < H_DIM; ++j) {
            float hv = h32[j];
            hv = hv > 0.f ? hv : 0.f;
            acc += hv * W_lin[j];
        }
        out[0] = acc;
    }
}

extern "C" void kernel_launch(void* const* d_in, const int* in_sizes, int n_in,
                              void* d_out, int out_size, void* d_ws, size_t ws_size,
                              hipStream_t stream) {
    const float* x     = (const float*)d_in[0];
    const float* W_ih  = (const float*)d_in[1];
    const float* W_hh  = (const float*)d_in[2];
    const float* b_ih  = (const float*)d_in[3];
    const float* b_hh  = (const float*)d_in[4];
    const float* W_lin = (const float*)d_in[5];
    const float* b_lin = (const float*)d_in[6];

    lstm_persistent_wmma<<<dim3(1), dim3(NTHREADS), 0, stream>>>(
        x, W_ih, W_hh, b_ih, b_hh, W_lin, b_lin, (float*)d_out);
}
